// TRANSFORMER_SINDY_38130719654540
// MI455X (gfx1250) — matrix operand order repaired and checked by reading, not compiled, not run on hardware
//
#include <hip/hip_runtime.h>
#include <hip/hip_bf16.h>
#include <math.h>

// ---------------- problem constants ----------------
#define Bsz   32
#define Ssz   500
#define DMsz  128
#define Hsz   64
#define Gsz   192            // 3*H
#define FFsz  128
#define NLsz  2
#define Mrows (Bsz*Ssz)      // 16000 tokens
#define LIBSZ 2145           // 1 + 64 + 2080
#define NTRI  2080
#define KPAD  2176           // LIB rounded up to 32

typedef __attribute__((ext_vector_type(16))) __bf16 v16bf;
typedef __attribute__((ext_vector_type(8)))  float  v8f;

// ---------------- helpers ----------------
__global__ void f2bf_kernel(const float* __restrict__ in, __bf16* __restrict__ out, int n) {
    int i = blockIdx.x * blockDim.x + threadIdx.x;
    if (i < n) out[i] = (__bf16)in[i];
}

__global__ void tri_init_kernel(int* __restrict__ ti, int* __restrict__ tj) {
    int q = blockIdx.x * blockDim.x + threadIdx.x;
    if (q >= NTRI) return;
    int i = 0, rem = q, len = Hsz;               // torch order: i<=j
    while (rem >= len) { rem -= len; ++i; --len; }
    ti[q] = i; tj[q] = i + rem;
}

// coef (NL, LIB, H) f32  ->  coefT (NL, H, KPAD) bf16, zero-padded in K
__global__ void coefT_kernel(const float* __restrict__ coef, __bf16* __restrict__ out) {
    int idx = blockIdx.x * blockDim.x + threadIdx.x;
    const int total = NLsz * Hsz * KPAD;
    if (idx >= total) return;
    int k = idx % KPAD; int t = idx / KPAD; int n = t % Hsz; int l = t / Hsz;
    out[idx] = (k < LIBSZ) ? (__bf16)coef[((size_t)l * LIBSZ + k) * Hsz + n] : (__bf16)0.f;
}

// ---- fragment loaders (layouts per CDNA5 ISA §7.12.2) ----
__device__ __forceinline__ v16bf load_afrag(const __bf16* __restrict__ arow, int k0, int hi) {
    v16bf a;
#pragma unroll
    for (int e = 0; e < 16; ++e) {
        int vg = e >> 1;
        // 16-bit A 16x32: vgpr<4 -> K 0..7 (+8*hi), vgpr>=4 -> K 16..23 (+8*hi)
        int ka = k0 + (vg < 4 ? 2 * vg : 2 * vg + 8) + 8 * hi + (e & 1);
        a[e] = arow[ka];
    }
    return a;
}
__device__ __forceinline__ v16bf load_bfrag(const __bf16* __restrict__ brow, int k0) {
    v16bf b;
#pragma unroll
    for (int e = 0; e < 16; ++e) b[e] = brow[k0 + (e < 8 ? e : e + 8)];   // K = k0 + 8*hi + e (e<8), k0 + 16 + 8*hi + (e-8) (e>=8) (hi folded into brow)
    return b;
}

// ---------- generic WMMA GEMM: C(M,N) = A(M,K) @ B(N,K)^T + bias ----------
// one wave -> 16x64 strip (4 WMMA tiles, A-frag reused 4x), software-pipelined K loop.
// N%64==0, K%32==0 (K>=64), M%16==0.
__global__ void wmma_gemm_nt(const __bf16* __restrict__ A, const __bf16* __restrict__ B,
                             const float* __restrict__ bias,
                             float* __restrict__ Cf, __bf16* __restrict__ Cbf,
                             int M, int N, int K, int do_gelu) {
    int lane = threadIdx.x & 31;
    int wave = threadIdx.x >> 5;
    int tilesNs = N >> 6, tilesM = M >> 4;
    int id = blockIdx.x * 8 + wave;
    if (id >= tilesM * tilesNs) return;
    int tm = id / tilesNs, ts = id - tm * tilesNs;
    int m0 = tm << 4, n0 = ts << 6;
    int lo = lane & 15, hi = lane >> 4;

    const __bf16* arow = A + (size_t)(m0 + lo) * K;
    const __bf16* brow = B + (size_t)(n0 + lo) * K + 8 * hi;

    v8f acc[4] = {};
    // prologue: fragments for k0 = 0
    v16bf a = load_afrag(arow, 0, hi);
    v16bf b[4];
#pragma unroll
    for (int s = 0; s < 4; ++s) b[s] = load_bfrag(brow + (size_t)(s * 16) * K, 0);

    int k0 = 0;
    for (; k0 + 32 < K; k0 += 32) {
        // issue next iteration's loads BEFORE this iteration's WMMAs
        v16bf an = load_afrag(arow, k0 + 32, hi);
        v16bf bn[4];
#pragma unroll
        for (int s = 0; s < 4; ++s) bn[s] = load_bfrag(brow + (size_t)(s * 16) * K, k0 + 32);
#pragma unroll
        for (int s = 0; s < 4; ++s)
            acc[s] = __builtin_amdgcn_wmma_f32_16x16x32_bf16(false, a, false, b[s],
                                                             (short)0, acc[s], false, false);
        a = an;
#pragma unroll
        for (int s = 0; s < 4; ++s) b[s] = bn[s];
    }
#pragma unroll
    for (int s = 0; s < 4; ++s)
        acc[s] = __builtin_amdgcn_wmma_f32_16x16x32_bf16(false, a, false, b[s],
                                                         (short)0, acc[s], false, false);

#pragma unroll
    for (int s = 0; s < 4; ++s) {
        int n = n0 + s * 16 + lo;
        float bval = bias ? bias[n] : 0.f;
#pragma unroll
        for (int v = 0; v < 8; ++v) {
            int m = m0 + v + 8 * hi;                            // C/D: VGPR v -> M = v + 8*hi
            float val = acc[s][v] + bval;
            if (do_gelu) val = 0.5f * val * (1.f + erff(val * 0.70710678118654752f));
            if (Cf)  Cf[(size_t)m * N + n]  = val;
            if (Cbf) Cbf[(size_t)m * N + n] = (__bf16)val;
        }
    }
}

// ---------- SINDy WMMA GEMM with on-the-fly Theta, z panel staged in LDS ----------
__device__ __forceinline__ v16bf theta_frag(const float* __restrict__ zr,
                                            const unsigned char* __restrict__ tish,
                                            const unsigned char* __restrict__ tjsh,
                                            int k0, int hi) {
    v16bf a;
#pragma unroll
    for (int e = 0; e < 16; ++e) {
        int vg = e >> 1;
        int ka = k0 + (vg < 4 ? 2 * vg : 2 * vg + 8) + 8 * hi + (e & 1);
        float th;
        if (ka >= LIBSZ)        th = 0.f;
        else if (ka == 0)       th = 1.f;
        else if (ka < 1 + Hsz)  th = zr[ka - 1];
        else { int q = ka - 1 - Hsz; th = zr[tish[q]] * zr[tjsh[q]]; }
        a[e] = (__bf16)th;
    }
    return a;
}

// upd(M,64) = Theta(Z)(M,LIB) @ coefT(64,KPAD)^T. One block = 8 M-tiles (128 rows).
// M multiple of 128 (16000 = 125*128).
__global__ void sindy_wmma_kernel(const float* __restrict__ Z, const __bf16* __restrict__ coefT,
                                  const int* __restrict__ ti, const int* __restrict__ tj,
                                  float* __restrict__ upd) {
    __shared__ float zsh[128 * 65];                             // stride 65: conflict-free
    __shared__ unsigned char tish[NTRI];
    __shared__ unsigned char tjsh[NTRI];
    int tid = threadIdx.x;
    int lane = tid & 31, wave = tid >> 5;
    int mbase = blockIdx.x * 128;

    for (int idx = tid; idx < 128 * 64; idx += 256) {
        int r = idx >> 6, c = idx & 63;
        zsh[r * 65 + c] = Z[(size_t)(mbase + r) * Hsz + c];
    }
    for (int idx = tid; idx < NTRI; idx += 256) {
        tish[idx] = (unsigned char)ti[idx];
        tjsh[idx] = (unsigned char)tj[idx];
    }
    __syncthreads();

    int lo = lane & 15, hi = lane >> 4;
    const float* zr = &zsh[(wave * 16 + lo) * 65];              // this lane's z row
    const __bf16* brow = coefT + (size_t)lo * KPAD + 8 * hi;

    v8f acc[4] = {};
    v16bf a = theta_frag(zr, tish, tjsh, 0, hi);
    v16bf b[4];
#pragma unroll
    for (int s = 0; s < 4; ++s) b[s] = load_bfrag(brow + (size_t)(s * 16) * KPAD, 0);

    int k0 = 0;
    for (; k0 + 32 < KPAD; k0 += 32) {
        v16bf an = theta_frag(zr, tish, tjsh, k0 + 32, hi);     // next Theta (LDS+VALU)
        v16bf bn[4];
#pragma unroll
        for (int s = 0; s < 4; ++s) bn[s] = load_bfrag(brow + (size_t)(s * 16) * KPAD, k0 + 32);
#pragma unroll
        for (int s = 0; s < 4; ++s)
            acc[s] = __builtin_amdgcn_wmma_f32_16x16x32_bf16(false, a, false, b[s],
                                                             (short)0, acc[s], false, false);
        a = an;
#pragma unroll
        for (int s = 0; s < 4; ++s) b[s] = bn[s];
    }
#pragma unroll
    for (int s = 0; s < 4; ++s)
        acc[s] = __builtin_amdgcn_wmma_f32_16x16x32_bf16(false, a, false, b[s],
                                                         (short)0, acc[s], false, false);

#pragma unroll
    for (int s = 0; s < 4; ++s) {
        int n = s * 16 + lo;
#pragma unroll
        for (int v = 0; v < 8; ++v) {
            int m = mbase + wave * 16 + v + 8 * hi;
            upd[(size_t)m * Hsz + n] = acc[s][v];
        }
    }
}

// ---------- GRU recurrence: one block per batch, 192 threads (one per gate row) ----------
__global__ void gru_scan_kernel(const float* __restrict__ xp,   // (B,S,192) = x@Wih^T + bih
                                const float* __restrict__ Whh,  // (192,64)
                                const float* __restrict__ bhh,  // (192)
                                float* __restrict__ Hout,       // (B,S,64)
                                __bf16* __restrict__ Hbf) {     // optional bf16 mirror
    __shared__ float Wl[Gsz * 65];                              // stride 65: no bank conflicts
    __shared__ float h[Hsz];
    __shared__ float gg[Gsz];
    int b = blockIdx.x, id = threadIdx.x;                       // 192 threads
    for (int idx = id; idx < Gsz * Hsz; idx += Gsz) {
        int r = idx >> 6, c = idx & 63;
        Wl[r * 65 + c] = Whh[idx];
    }
    if (id < Hsz) h[id] = 0.f;
    float bg = bhh[id];
    __syncthreads();

    const float* w = &Wl[id * 65];
    const float* xpb = xp + (size_t)b * Ssz * Gsz;
    float* hob = Hout + (size_t)b * Ssz * Hsz;

    for (int t = 0; t < Ssz; ++t) {
        float g = bg;
#pragma unroll 8
        for (int k = 0; k < Hsz; ++k) g += h[k] * w[k];
        gg[id] = g;
        __syncthreads();
        float hn = 0.f;
        if (id < Hsz) {
            const float* xt = xpb + (size_t)t * Gsz;
            float r  = 1.f / (1.f + __expf(-(xt[id] + gg[id])));
            float zg = 1.f / (1.f + __expf(-(xt[Hsz + id] + gg[Hsz + id])));
            float ng = tanhf(xt[2 * Hsz + id] + r * gg[2 * Hsz + id]);
            hn = (1.f - zg) * ng + zg * h[id];
        }
        __syncthreads();                                        // all dot-reads of h done
        if (id < Hsz) {
            h[id] = hn;
            hob[(size_t)t * Hsz + id] = hn;
            if (Hbf) Hbf[(size_t)b * Ssz * Hsz + (size_t)t * Hsz + id] = (__bf16)hn;
        }
        __syncthreads();
    }
}

// ---------- residual add + LayerNorm (H=64): one wave per row ----------
__global__ void add_ln_kernel(const float* __restrict__ X, const float* __restrict__ Y,
                              const float* __restrict__ g, const float* __restrict__ bta,
                              float* __restrict__ Out, __bf16* __restrict__ Obf, int rows) {
    int lane = threadIdx.x & 31;
    int row = blockIdx.x * (blockDim.x >> 5) + (threadIdx.x >> 5);
    if (row >= rows) return;
    const float* xr = X + (size_t)row * Hsz;
    const float* yr = Y + (size_t)row * Hsz;
    float v0 = xr[lane] + yr[lane];
    float v1 = xr[lane + 32] + yr[lane + 32];
    float s = v0 + v1;
    for (int m = 16; m >= 1; m >>= 1) s += __shfl_xor(s, m, 32);
    float mean = s * (1.f / 64.f);
    float d0 = v0 - mean, d1 = v1 - mean;
    float vs = d0 * d0 + d1 * d1;
    for (int m = 16; m >= 1; m >>= 1) vs += __shfl_xor(vs, m, 32);
    float rstd = rsqrtf(vs * (1.f / 64.f) + 1e-5f);
    float o0 = d0 * rstd * g[lane] + bta[lane];
    float o1 = d1 * rstd * g[lane + 32] + bta[lane + 32];
    Out[(size_t)row * Hsz + lane] = o0;
    Out[(size_t)row * Hsz + lane + 32] = o1;
    if (Obf) {
        Obf[(size_t)row * Hsz + lane] = (__bf16)o0;
        Obf[(size_t)row * Hsz + lane + 32] = (__bf16)o1;
    }
}

// ---------- final output: full h + last timestep ----------
__global__ void finalize_kernel(const float* __restrict__ H, float* __restrict__ out) {
    int idx = blockIdx.x * blockDim.x + threadIdx.x;
    const int total = Mrows * Hsz;
    if (idx < total) out[idx] = H[idx];
    if (idx < Bsz * Hsz) {
        int b = idx >> 6, i = idx & 63;
        out[total + idx] = H[((size_t)b * Ssz + (Ssz - 1)) * Hsz + i];
    }
}

// ---------------- host launch ----------------
static inline size_t alignup(size_t x) { return (x + 255) & ~(size_t)255; }

extern "C" void kernel_launch(void* const* d_in, const int* in_sizes, int n_in,
                              void* d_out, int out_size, void* d_ws, size_t ws_size,
                              hipStream_t stream) {
    (void)in_sizes; (void)n_in; (void)out_size; (void)ws_size;
    const float* x     = (const float*)d_in[0];
    const float* Wih0  = (const float*)d_in[1];
    const float* Whh0  = (const float*)d_in[2];
    const float* bih0  = (const float*)d_in[3];
    const float* bhh0  = (const float*)d_in[4];
    const float* Wih1  = (const float*)d_in[5];
    const float* Whh1  = (const float*)d_in[6];
    const float* bih1  = (const float*)d_in[7];
    const float* bhh1  = (const float*)d_in[8];
    const float* coeffs= (const float*)d_in[9];   // (NL, LIB, H)
    const float* ln1g  = (const float*)d_in[10];
    const float* ln1b  = (const float*)d_in[11];
    const float* W1    = (const float*)d_in[12];  // (NL, FF, H)
    const float* b1    = (const float*)d_in[13];
    const float* W2    = (const float*)d_in[14];  // (NL, H, FF)
    const float* b2    = (const float*)d_in[15];
    const float* ln2g  = (const float*)d_in[16];
    const float* ln2b  = (const float*)d_in[17];

    char* ws = (char*)d_ws; size_t off = 0;
    auto carve = [&](size_t bytes) { void* p = ws + off; off += alignup(bytes); return p; };

    __bf16* x_bf    = (__bf16*)carve((size_t)Mrows * DMsz * 2);
    float*  xp      = (float*) carve((size_t)Mrows * Gsz  * 4);
    float*  h0      = (float*) carve((size_t)Mrows * Hsz  * 4);
    __bf16* h0_bf   = (__bf16*)carve((size_t)Mrows * Hsz  * 2);
    float*  h       = (float*) carve((size_t)Mrows * Hsz  * 4);
    float*  upd     = (float*) carve((size_t)Mrows * Hsz  * 4);
    float*  src     = (float*) carve((size_t)Mrows * Hsz  * 4);
    __bf16* src_bf  = (__bf16*)carve((size_t)Mrows * Hsz  * 2);
    __bf16* ff1_bf  = (__bf16*)carve((size_t)Mrows * FFsz * 2);
    float*  ff2     = (float*) carve((size_t)Mrows * Hsz  * 4);
    __bf16* wih0_bf = (__bf16*)carve((size_t)Gsz * DMsz * 2);
    __bf16* wih1_bf = (__bf16*)carve((size_t)Gsz * Hsz  * 2);
    __bf16* coefT   = (__bf16*)carve((size_t)NLsz * Hsz * KPAD * 2);
    __bf16* w1_bf   = (__bf16*)carve((size_t)NLsz * FFsz * Hsz * 2);
    __bf16* w2_bf   = (__bf16*)carve((size_t)NLsz * Hsz * FFsz * 2);
    int*    tri_i   = (int*)   carve((size_t)NTRI * 4);
    int*    tri_j   = (int*)   carve((size_t)NTRI * 4);

    auto cvt = [&](const float* in, __bf16* out, int n) {
        f2bf_kernel<<<(n + 255) / 256, 256, 0, stream>>>(in, out, n);
    };
    cvt(x, x_bf, Mrows * DMsz);
    cvt(Wih0, wih0_bf, Gsz * DMsz);
    cvt(Wih1, wih1_bf, Gsz * Hsz);
    cvt(W1, w1_bf, NLsz * FFsz * Hsz);
    cvt(W2, w2_bf, NLsz * Hsz * FFsz);
    {
        int n = NLsz * Hsz * KPAD;
        coefT_kernel<<<(n + 255) / 256, 256, 0, stream>>>(coeffs, coefT);
    }
    tri_init_kernel<<<(NTRI + 255) / 256, 256, 0, stream>>>(tri_i, tri_j);

    auto gemm = [&](const __bf16* A, const __bf16* B, const float* bias,
                    float* Cf, __bf16* Cbf, int M, int N, int K, int gelu) {
        int ids = (M >> 4) * (N >> 6);
        wmma_gemm_nt<<<(ids + 7) / 8, 256, 0, stream>>>(A, B, bias, Cf, Cbf, M, N, K, gelu);
    };

    // GRU layer 0
    gemm(x_bf, wih0_bf, bih0, xp, nullptr, Mrows, Gsz, DMsz, 0);
    gru_scan_kernel<<<Bsz, Gsz, 0, stream>>>(xp, Whh0, bhh0, h0, h0_bf);
    // GRU layer 1
    gemm(h0_bf, wih1_bf, bih1, xp, nullptr, Mrows, Gsz, Hsz, 0);
    gru_scan_kernel<<<Bsz, Gsz, 0, stream>>>(xp, Whh1, bhh1, h, nullptr);

    // SINDy transformer layers
    for (int l = 0; l < NLsz; ++l) {
        sindy_wmma_kernel<<<Mrows / 128, 256, 0, stream>>>(
            h, coefT + (size_t)l * Hsz * KPAD, tri_i, tri_j, upd);
        add_ln_kernel<<<(Mrows + 7) / 8, 256, 0, stream>>>(
            h, upd, ln1g + l * Hsz, ln1b + l * Hsz, src, src_bf, Mrows);
        gemm(src_bf, w1_bf + (size_t)l * FFsz * Hsz, b1 + l * FFsz,
             nullptr, ff1_bf, Mrows, FFsz, Hsz, 1);                    // GELU fused
        gemm(ff1_bf, w2_bf + (size_t)l * Hsz * FFsz, b2 + l * Hsz,
             ff2, nullptr, Mrows, Hsz, FFsz, 0);
        add_ln_kernel<<<(Mrows + 7) / 8, 256, 0, stream>>>(
            src, ff2, ln2g + l * Hsz, ln2b + l * Hsz, h, nullptr, Mrows);
    }

    finalize_kernel<<<(Mrows * Hsz + 255) / 256, 256, 0, stream>>>(h, (float*)d_out);
}
